// AdvancedTransformerLayer_53326313947857
// MI455X (gfx1250) — matrix operand order repaired and checked by reading, not compiled, not run on hardware
//
#include <hip/hip_runtime.h>

#define Bsz 4
#define Ssz 2048
#define Dsz 1024
#define Hsz 16
#define Fsz 4096
#define DHsz 64
#define BSsz 64

typedef __attribute__((ext_vector_type(16))) __bf16 v16bf;
typedef __attribute__((ext_vector_type(8)))  float  v8f;
typedef int v4i __attribute__((vector_size(4 * sizeof(int))));

union Frag {
    unsigned int u[8];
    uint4 q[2];
    v16bf v;
};

__device__ __forceinline__ unsigned short f2bf(float f) {
    unsigned int u = __float_as_uint(f);
    return (unsigned short)((u + 0x7FFFu + ((u >> 16) & 1u)) >> 16);
}

// ---------------------------------------------------------------------------
// Async global->LDS copy (CDNA5 GLOBAL_LOAD_ASYNC_TO_LDS_B128, ASYNCcnt) with
// compile-time fallback to plain load+store if the builtin is unavailable.
// Signature (from clang diagnostic): arg0 = v4i __device__* (AS1), 4 args.
// ---------------------------------------------------------------------------
#if __has_builtin(__builtin_amdgcn_global_load_async_to_lds_b128) && \
    __has_builtin(__builtin_amdgcn_s_wait_asynccnt)
__device__ __forceinline__ void async_cp16(const void* g, void* l) {
    __builtin_amdgcn_global_load_async_to_lds_b128(
        (__attribute__((address_space(1))) v4i*)g,
        (__attribute__((address_space(3))) v4i*)l, 0, 0);
}
__device__ __forceinline__ void async_wait() {
    __builtin_amdgcn_s_wait_asynccnt(0);
}
#else
__device__ __forceinline__ void async_cp16(const void* g, void* l) {
    *(uint4*)l = *(const uint4*)g;
}
__device__ __forceinline__ void async_wait() {}
#endif

// ---------------------------------------------------------------------------
// fp32 -> bf16 conversion (weights)
// ---------------------------------------------------------------------------
__global__ __launch_bounds__(256) void cvt_bf16_kernel(
    const float* __restrict__ in, unsigned short* __restrict__ out, int n) {
    int i = blockIdx.x * 256 + threadIdx.x;
    if (i < n) out[i] = f2bf(in[i]);
}

// ---------------------------------------------------------------------------
// LayerNorm over D=1024, one block (256 thr) per row, bf16 output
// ---------------------------------------------------------------------------
__global__ __launch_bounds__(256) void ln_bf16_kernel(
    const float* __restrict__ x, const float* __restrict__ g,
    const float* __restrict__ b, unsigned short* __restrict__ out) {
    __shared__ float r1[256];
    __shared__ float r2[256];
    int tid = threadIdx.x;
    size_t row = blockIdx.x;
    float4 v = ((const float4*)(x + row * Dsz))[tid];
    float s  = v.x + v.y + v.z + v.w;
    float ss = v.x * v.x + v.y * v.y + v.z * v.z + v.w * v.w;
    r1[tid] = s; r2[tid] = ss;
    __syncthreads();
    for (int o = 128; o > 0; o >>= 1) {
        if (tid < o) { r1[tid] += r1[tid + o]; r2[tid] += r2[tid + o]; }
        __syncthreads();
    }
    float mean = r1[0] * (1.0f / Dsz);
    float var  = r2[0] * (1.0f / Dsz) - mean * mean;
    float rstd = rsqrtf(var + 1e-5f);
    float4 gv = ((const float4*)g)[tid];
    float4 bv = ((const float4*)b)[tid];
    unsigned short* o4 = out + row * Dsz + tid * 4;
    o4[0] = f2bf((v.x - mean) * rstd * gv.x + bv.x);
    o4[1] = f2bf((v.y - mean) * rstd * gv.y + bv.y);
    o4[2] = f2bf((v.z - mean) * rstd * gv.z + bv.z);
    o4[3] = f2bf((v.w - mean) * rstd * gv.w + bv.w);
}

// ---------------------------------------------------------------------------
// Tiled WMMA GEMM: C[M,N] = A[M,K](bf16) * B[K,N](bf16) + bias, epilogues:
//   EPI=0: store bf16 into blocked head layout [b][h][s/64][64][64] (QKV)
//   EPI=1: exact GELU, store bf16 row-major                         (FFN1)
//   EPI=2: + resid (fp32), store fp32 row-major                     (O/FFN2)
// Tile: 128x128, 256 threads = 8 waves, each wave -> 16 rows x 128 cols.
// LDS layouts are fragment-major: each lane's WMMA operand = 2x ds_load_b128.
// ---------------------------------------------------------------------------
#define ASTR 20   // A row stride in dwords (16 data + pad), 16B-aligned groups
#define BSTR 20   // B col stride in dwords (16 k-pairs + pad)

template <int EPI>
__global__ __launch_bounds__(256) void wmma_gemm_kernel(
    const unsigned short* __restrict__ A, const unsigned short* __restrict__ Bw,
    const float* __restrict__ bias, const float* __restrict__ resid,
    float* __restrict__ Cf, unsigned short* __restrict__ Cb,
    int M, int N, int K) {
    __shared__ unsigned int a_lds[128 * ASTR];   // [row][k-pair]
    __shared__ unsigned int b_lds[128 * BSTR];   // [col][k-pair] (transposed)
    int tid = threadIdx.x, lane = tid & 31, wave = tid >> 5;
    int half = lane >> 4, lc = lane & 15;
    int tile_m = blockIdx.y * 128, tile_n = blockIdx.x * 128;

    v8f acc[8] = {};
    for (int k0 = 0; k0 < K; k0 += 32) {
        // stage A tile (128 x 32 bf16): direct 16B copies (async when available)
        for (int t = tid; t < 512; t += 256) {
            int r = t >> 2, c4 = (t & 3) * 4;
            async_cp16(A + (size_t)(tile_m + r) * K + k0 + c4 * 2,
                       &a_lds[r * ASTR + c4]);
        }
        // stage B tile (32 x 128) into [col][pair]: dword = {B[2p][n],B[2p+1][n]}
        for (int t = tid; t < 512; t += 256) {
            int p = t >> 5, c = (t & 31) * 4;
            const unsigned short* s0 = Bw + (size_t)(k0 + 2 * p) * N + tile_n + c;
            ushort4 r0 = *(const ushort4*)s0;
            ushort4 r1 = *(const ushort4*)(s0 + N);
            b_lds[(c + 0) * BSTR + p] = (unsigned)r0.x | ((unsigned)r1.x << 16);
            b_lds[(c + 1) * BSTR + p] = (unsigned)r0.y | ((unsigned)r1.y << 16);
            b_lds[(c + 2) * BSTR + p] = (unsigned)r0.z | ((unsigned)r1.z << 16);
            b_lds[(c + 3) * BSTR + p] = (unsigned)r0.w | ((unsigned)r1.w << 16);
        }
        if (k0 + 32 < K)
            __builtin_prefetch(A + (size_t)(tile_m + (tid & 127)) * K + k0 + 32, 0, 0);
        async_wait();
        __syncthreads();

        // A fragment: two contiguous 16B groups per lane
        Frag a;
        {
            const unsigned int* ar = &a_lds[(wave * 16 + lc) * ASTR];
            a.q[0] = *(const uint4*)(ar + half * 4);
            a.q[1] = *(const uint4*)(ar + 8 + half * 4);
        }
        #pragma unroll
        for (int j = 0; j < 8; ++j) {
            Frag b;
            const unsigned int* bp = &b_lds[(j * 16 + lc) * BSTR + half * 4];
            b.q[0] = *(const uint4*)(bp);
            b.q[1] = *(const uint4*)(bp + 8);
            acc[j] = __builtin_amdgcn_wmma_f32_16x16x32_bf16(
                false, a.v, false, b.v, (short)0, acc[j], false, false);
        }
        __syncthreads();
    }

    // epilogue
    #pragma unroll
    for (int j = 0; j < 8; ++j) {
        int gn = tile_n + j * 16 + lc;
        float bv = bias[gn];
        #pragma unroll
        for (int r = 0; r < 8; ++r) {
            int gm = tile_m + wave * 16 + half * 8 + r;
            float val = acc[j][r] + bv;
            if (EPI == 2) {
                Cf[(size_t)gm * N + gn] = val + resid[(size_t)gm * N + gn];
            } else if (EPI == 1) {
                val = 0.5f * val * (1.0f + erff(val * 0.70710678118654752f));
                Cb[(size_t)gm * N + gn] = f2bf(val);
            } else {
                int bb = gm >> 11, s = gm & (Ssz - 1);
                int h = gn >> 6, dh = gn & 63;
                size_t oi = ((((size_t)bb * Hsz + h) * (Ssz / BSsz) + (s >> 6)) * 64 + (s & 63)) * 64 + dh;
                Cb[oi] = f2bf(val);
            }
        }
    }
}

// ---------------------------------------------------------------------------
// Block-sparse attention: one block per (b,h,seq-block). 128 thr = 4 waves.
// scores = Q K^T / 8 (WMMA), fp32 softmax (probs -> d_out), O = P V (WMMA).
// O written bf16 in [b][s][h][dh] layout (= GEMM A matrix for O-projection).
// All operand tiles in fragment-major LDS layout (2x b128 per fragment).
// ---------------------------------------------------------------------------
#define QSTR 36   // 32 data dwords + pad, keeps 16B group alignment

__global__ __launch_bounds__(128) void attn_kernel(
    const unsigned short* __restrict__ qb, const unsigned short* __restrict__ kb,
    const unsigned short* __restrict__ vb, float* __restrict__ attnw,
    unsigned short* __restrict__ o_out) {
    __shared__ unsigned int q_lds[64 * QSTR];   // [query][dh-pair]
    __shared__ unsigned int k_lds[64 * QSTR];   // [key][dh-pair]   (B frag rows)
    __shared__ unsigned int v_lds[64 * QSTR];   // [dh][key-pair]   (B frag rows)
    __shared__ float        s_lds[64 * 65];     // scores fp32
    __shared__ unsigned int p_lds[64 * QSTR];   // [query][key-pair] probs bf16

    int tid = threadIdx.x, lane = tid & 31, wave = tid >> 5;
    int half = lane >> 4, lc = lane & 15;
    int blk = blockIdx.x;                       // ((b*H+h)*32+n)
    size_t base = (size_t)blk * (BSsz * DHsz);

    // stage Q and K: rows are already pair-major along dh -> direct 16B copies
    for (int t = tid; t < 512; t += 128) {
        int r = t >> 3, c4 = (t & 7) * 4;
        async_cp16(qb + base + r * 64 + c4 * 2, &q_lds[r * QSTR + c4]);
        async_cp16(kb + base + r * 64 + c4 * 2, &k_lds[r * QSTR + c4]);
    }
    // stage V packed into [dh][key-pair]: dword = {V[2p][dh], V[2p+1][dh]}
    for (int t = tid; t < 512; t += 128) {
        int p = t >> 4, c = (t & 15) * 4;
        ushort4 a = *(const ushort4*)(vb + base + (2 * p) * 64 + c);
        ushort4 b = *(const ushort4*)(vb + base + (2 * p + 1) * 64 + c);
        v_lds[(c + 0) * QSTR + p] = (unsigned)a.x | ((unsigned)b.x << 16);
        v_lds[(c + 1) * QSTR + p] = (unsigned)a.y | ((unsigned)b.y << 16);
        v_lds[(c + 2) * QSTR + p] = (unsigned)a.z | ((unsigned)b.z << 16);
        v_lds[(c + 3) * QSTR + p] = (unsigned)a.w | ((unsigned)b.w << 16);
    }
    async_wait();
    __syncthreads();

    // scores: each wave -> 16 query rows x 64 keys, K-dim = 64 (2 chunks)
    v8f sa[4] = {};
    {
        int row = wave * 16 + lc;
        #pragma unroll
        for (int ch = 0; ch < 2; ++ch) {
            Frag a;
            const unsigned int* qr = &q_lds[row * QSTR + ch * 16];
            a.q[0] = *(const uint4*)(qr + half * 4);
            a.q[1] = *(const uint4*)(qr + 8 + half * 4);
            #pragma unroll
            for (int j = 0; j < 4; ++j) {
                Frag b;
                const unsigned int* kp = &k_lds[(j * 16 + lc) * QSTR + ch * 16 + half * 4];
                b.q[0] = *(const uint4*)(kp);
                b.q[1] = *(const uint4*)(kp + 8);
                sa[j] = __builtin_amdgcn_wmma_f32_16x16x32_bf16(
                    false, a.v, false, b.v, (short)0, sa[j], false, false);
            }
        }
    }
    #pragma unroll
    for (int j = 0; j < 4; ++j)
        #pragma unroll
        for (int r = 0; r < 8; ++r)
            s_lds[(wave * 16 + half * 8 + r) * 65 + j * 16 + lc] = sa[j][r] * 0.125f;
    __syncthreads();

    // softmax: one thread per query row
    if (tid < 64) {
        float* sr = &s_lds[tid * 65];
        float m = sr[0];
        for (int c = 1; c < 64; ++c) m = fmaxf(m, sr[c]);
        float sum = 0.0f;
        for (int c = 0; c < 64; ++c) { float e = expf(sr[c] - m); sr[c] = e; sum += e; }
        float inv = 1.0f / sum;
        float* aw = attnw + base;  // blk * 4096
        for (int c2 = 0; c2 < 32; ++c2) {
            float p0 = sr[2 * c2] * inv, p1 = sr[2 * c2 + 1] * inv;
            aw[tid * 64 + 2 * c2]     = p0;
            aw[tid * 64 + 2 * c2 + 1] = p1;
            p_lds[tid * QSTR + c2] = (unsigned)f2bf(p0) | ((unsigned)f2bf(p1) << 16);
        }
    }
    __syncthreads();

    // O = P * V
    v8f oa[4] = {};
    {
        int row = wave * 16 + lc;
        #pragma unroll
        for (int ch = 0; ch < 2; ++ch) {
            Frag a;
            const unsigned int* pr = &p_lds[row * QSTR + ch * 16];
            a.q[0] = *(const uint4*)(pr + half * 4);
            a.q[1] = *(const uint4*)(pr + 8 + half * 4);
            #pragma unroll
            for (int j = 0; j < 4; ++j) {
                Frag b;
                const unsigned int* vp = &v_lds[(j * 16 + lc) * QSTR + ch * 16 + half * 4];
                b.q[0] = *(const uint4*)(vp);
                b.q[1] = *(const uint4*)(vp + 8);
                oa[j] = __builtin_amdgcn_wmma_f32_16x16x32_bf16(
                    false, a.v, false, b.v, (short)0, oa[j], false, false);
            }
        }
    }
    int n = blk & 31, bh = blk >> 5, h = bh & 15, bb = bh >> 4;
    #pragma unroll
    for (int j = 0; j < 4; ++j)
        #pragma unroll
        for (int r = 0; r < 8; ++r) {
            int s = n * 64 + wave * 16 + half * 8 + r;
            size_t oi = (((size_t)bb * Ssz + s) * Hsz + h) * DHsz + j * 16 + lc;
            o_out[oi] = f2bf(oa[j][r]);
        }
}

// ---------------------------------------------------------------------------
extern "C" void kernel_launch(void* const* d_in, const int* in_sizes, int n_in,
                              void* d_out, int out_size, void* d_ws, size_t ws_size,
                              hipStream_t stream) {
    const float* hidden = (const float*)d_in[0];
    const float* ln1_g = (const float*)d_in[1];
    const float* ln1_b = (const float*)d_in[2];
    const float* ln2_g = (const float*)d_in[3];
    const float* ln2_b = (const float*)d_in[4];
    const float* wq = (const float*)d_in[5];  const float* bq = (const float*)d_in[6];
    const float* wk = (const float*)d_in[7];  const float* bk = (const float*)d_in[8];
    const float* wv = (const float*)d_in[9];  const float* bv = (const float*)d_in[10];
    const float* wo = (const float*)d_in[11]; const float* bo = (const float*)d_in[12];
    const float* w1 = (const float*)d_in[13]; const float* b1 = (const float*)d_in[14];
    const float* w2 = (const float*)d_in[15]; const float* b2 = (const float*)d_in[16];

    const size_t M = (size_t)Bsz * Ssz;            // 8192
    const size_t nAct = M * Dsz;                   // 8.39M
    const size_t nFF  = M * Fsz;                   // 33.55M
    const size_t nWqkv = (size_t)Dsz * Dsz;        // 1.05M
    const size_t nW1  = (size_t)Dsz * Fsz;

    char* ws = (char*)d_ws;
    size_t off = 0;
    auto alloc = [&](size_t bytes) {
        void* p = ws + off;
        off += (bytes + 255) & ~(size_t)255;
        return p;
    };
    unsigned short* xb  = (unsigned short*)alloc(nAct * 2);   // LN output (reused LN1/LN2)
    unsigned short* wqb = (unsigned short*)alloc(nWqkv * 2);
    unsigned short* wkb = (unsigned short*)alloc(nWqkv * 2);
    unsigned short* wvb = (unsigned short*)alloc(nWqkv * 2);
    unsigned short* wob = (unsigned short*)alloc(nWqkv * 2);
    unsigned short* w1b = (unsigned short*)alloc(nW1 * 2);
    unsigned short* w2b = (unsigned short*)alloc(nW1 * 2);
    unsigned short* qbk = (unsigned short*)alloc(nAct * 2);
    unsigned short* kbk = (unsigned short*)alloc(nAct * 2);
    unsigned short* vbk = (unsigned short*)alloc(nAct * 2);
    unsigned short* ob  = (unsigned short*)alloc(nAct * 2);
    float*          hbuf = (float*)alloc(nAct * 4);
    unsigned short* ff1 = (unsigned short*)alloc(nFF * 2);

    // 1) weights -> bf16
    cvt_bf16_kernel<<<(int)((nWqkv + 255) / 256), 256, 0, stream>>>(wq, wqb, (int)nWqkv);
    cvt_bf16_kernel<<<(int)((nWqkv + 255) / 256), 256, 0, stream>>>(wk, wkb, (int)nWqkv);
    cvt_bf16_kernel<<<(int)((nWqkv + 255) / 256), 256, 0, stream>>>(wv, wvb, (int)nWqkv);
    cvt_bf16_kernel<<<(int)((nWqkv + 255) / 256), 256, 0, stream>>>(wo, wob, (int)nWqkv);
    cvt_bf16_kernel<<<(int)((nW1 + 255) / 256), 256, 0, stream>>>(w1, w1b, (int)nW1);
    cvt_bf16_kernel<<<(int)((nW1 + 255) / 256), 256, 0, stream>>>(w2, w2b, (int)nW1);

    // 2) LN1
    ln_bf16_kernel<<<(int)M, 256, 0, stream>>>(hidden, ln1_g, ln1_b, xb);

    // 3) Q/K/V projections -> blocked head layout
    dim3 gD(Dsz / 128, (int)(M / 128));
    wmma_gemm_kernel<0><<<gD, 256, 0, stream>>>(xb, wqb, bq, nullptr, nullptr, qbk, (int)M, Dsz, Dsz);
    wmma_gemm_kernel<0><<<gD, 256, 0, stream>>>(xb, wkb, bk, nullptr, nullptr, kbk, (int)M, Dsz, Dsz);
    wmma_gemm_kernel<0><<<gD, 256, 0, stream>>>(xb, wvb, bv, nullptr, nullptr, vbk, (int)M, Dsz, Dsz);

    // 4) block attention; probs -> second output region
    float* out1 = (float*)d_out;
    float* attnw = out1 + nAct;
    attn_kernel<<<Bsz * Hsz * (Ssz / BSsz), 128, 0, stream>>>(qbk, kbk, vbk, attnw, ob);

    // 5) O projection + residual -> h (fp32)
    wmma_gemm_kernel<2><<<gD, 256, 0, stream>>>(ob, wob, bo, hidden, hbuf, nullptr, (int)M, Dsz, Dsz);

    // 6) LN2
    ln_bf16_kernel<<<(int)M, 256, 0, stream>>>(hbuf, ln2_g, ln2_b, xb);

    // 7) FFN up + exact GELU -> bf16
    dim3 gF(Fsz / 128, (int)(M / 128));
    wmma_gemm_kernel<1><<<gF, 256, 0, stream>>>(xb, w1b, b1, nullptr, nullptr, ff1, (int)M, Fsz, Dsz);

    // 8) FFN down + residual h -> final output (fp32)
    wmma_gemm_kernel<2><<<gD, 256, 0, stream>>>(ff1, w2b, b2, hbuf, out1, nullptr, (int)M, Dsz, Fsz);
}
